// ConditionalFlowModel_16518444220982
// MI455X (gfx1250) — matrix operand order repaired and checked by reading, not compiled, not run on hardware
//
// ConditionalFlowModel for MI455X (gfx1250): bf16 WMMA GEMM backbone.
// Compute-bound workload (~440 GFLOP/call, >800 FLOP/byte) -> everything
// routed through v_wmma_f32_16x16x32_bf16 with LDS double buffering.
// This revision: BM=128 BN=128 tile, 8 WMMAs/K-step per wave with batched
// fragment loads (one dscnt wait per group), native bf16 cvt, packed u32
// LDS stores for the transposed B tile.
#include <hip/hip_runtime.h>
#include <hip/hip_bf16.h>
#include <math.h>

typedef __attribute__((ext_vector_type(16))) __bf16 v16bf;
typedef __attribute__((ext_vector_type(8)))  __bf16 v8bf;
typedef __attribute__((ext_vector_type(8)))  float  v8f;

#define DEVFN static __device__ __forceinline__

// ---------------------------------------------------------------- helpers
DEVFN __bf16 f2bf(float f) { return (__bf16)f; }   // let backend pick hw cvt
DEVFN unsigned pack_bf2(float lo, float hi) {
  unsigned short a = __builtin_bit_cast(unsigned short, f2bf(lo));
  unsigned short b = __builtin_bit_cast(unsigned short, f2bf(hi));
  return (unsigned)a | ((unsigned)b << 16);
}
DEVFN float gelu_exact(float x) { return 0.5f * x * (1.0f + erff(x * 0.7071067811865475f)); }

DEVFN v16bf frag_cat(const __bf16* p0, const __bf16* p1) {
  v8bf a = *(const v8bf*)p0;        // 16B aligned by construction
  v8bf b = *(const v8bf*)p1;
  v16bf r;
#pragma unroll
  for (int i = 0; i < 8; ++i) { r[i] = a[i]; r[i + 8] = b[i]; }
  return r;
}

// ---------------------------------------------------------------- GEMM
// C[M,N] = act(A[M,K] @ W[K,N] + bias) + add   (all f32 in memory, bf16 math)
struct GemmArgs {
  const float* A; const float* W; const float* bias; const float* add;
  float* C;
  int M, N, K, lda, ldw, ldc, act;  // act: 0 none, 1 relu, 2 gelu(exact)
};

// BM=128 BN=128 BK=32, 256 threads = 8 wave32 in a 4x2 grid; each wave does
// a 32x64 strip = 2x4 wmma accumulators, 8 v_wmma per K-step with shared
// A/B fragments. LDS rows padded to 40 bf16 (80B): 16B-aligned fragments,
// bank-conflict-free (gcd(20,64)=4 -> 16 distinct banks across lanes).
__global__ __launch_bounds__(256) void gemm_bf16_wmma(GemmArgs g) {
  __shared__ __bf16 sA[2][128 * 40];
  __shared__ __bf16 sB[2][128 * 40];

  const int tid = threadIdx.x;
  const int wid = tid >> 5, lane = tid & 31;
  const int lm = lane & 15, kh = lane >> 4;
  const int wr = wid & 3, wc = wid >> 2;             // wave 4x2 grid
  const int nbase = blockIdx.x * 128, mbase = blockIdx.y * 128;

  float4 ra[4]; float4 rb0[2]; float4 rb1[2];

  auto fetch = [&](int k0) {
#pragma unroll
    for (int i = 0; i < 4; ++i) {          // A tile 128x32 f32, 4 float4/thread
      int f = tid + i * 256, r = f >> 3, c4 = f & 7;
      int row = mbase + r;
      if (row < g.M) {
        const float* p = g.A + (size_t)row * g.lda + k0 + c4 * 4;
        ra[i] = *(const float4*)p;
        if (k0 + 32 < g.K) __builtin_prefetch(p + 32, 0, 0);   // global_prefetch_b8
      } else ra[i] = make_float4(0.f, 0.f, 0.f, 0.f);
    }
#pragma unroll
    for (int i = 0; i < 2; ++i) {          // W tile 32x128 f32, k-pair x 4n units
      int f = tid + i * 256, kp = f >> 5, n4 = f & 31;   // k = 2*kp
      int col = nbase + n4 * 4;
      if (col < g.N) {
        const float* p = g.W + (size_t)(k0 + 2 * kp) * g.ldw + col;
        rb0[i] = *(const float4*)p;
        rb1[i] = *(const float4*)(p + g.ldw);
        if (k0 + 32 < g.K) __builtin_prefetch(p + (size_t)32 * g.ldw, 0, 0);
      } else { rb0[i] = make_float4(0.f, 0.f, 0.f, 0.f); rb1[i] = rb0[i]; }
    }
  };
  auto stage = [&](int buf) {
#pragma unroll
    for (int i = 0; i < 4; ++i) {
      int f = tid + i * 256, r = f >> 3, c4 = f & 7;
      __bf16* d = &sA[buf][r * 40 + c4 * 4];
      d[0] = f2bf(ra[i].x); d[1] = f2bf(ra[i].y); d[2] = f2bf(ra[i].z); d[3] = f2bf(ra[i].w);
    }
#pragma unroll
    for (int i = 0; i < 2; ++i) {          // store W transposed: sB[n][k], u32-packed k pairs
      int f = tid + i * 256, kp = f >> 5, n4 = f & 31;
      float l0[4] = {rb0[i].x, rb0[i].y, rb0[i].z, rb0[i].w};
      float l1[4] = {rb1[i].x, rb1[i].y, rb1[i].z, rb1[i].w};
#pragma unroll
      for (int j = 0; j < 4; ++j)
        *(unsigned*)&sB[buf][(n4 * 4 + j) * 40 + 2 * kp] = pack_bf2(l0[j], l1[j]);
    }
  };

  v8f zero = {};
  v8f acc[2][4];
#pragma unroll
  for (int mm = 0; mm < 2; ++mm)
#pragma unroll
    for (int nn = 0; nn < 4; ++nn) acc[mm][nn] = zero;

  fetch(0);
  stage(0);
  __syncthreads();

  const int nk = g.K >> 5;
  int cur = 0;
  for (int kt = 0; kt < nk; ++kt) {
    if (kt + 1 < nk) fetch((kt + 1) << 5);

    const __bf16* As = sA[cur];
    const __bf16* Bs = sB[cur];
    // Batch ALL fragment loads, then issue 8 wmma back-to-back (single
    // dscnt wait instead of one per wmma).
    v16bf af[2];
#pragma unroll
    for (int mm = 0; mm < 2; ++mm) {       // lane holds row m, K = kh*8..+7, 16+kh*8..+7
      const int am = (wr << 5) + (mm << 4) + lm;
      af[mm] = frag_cat(As + am * 40 + kh * 8, As + am * 40 + 16 + kh * 8);
    }
    v16bf bfg[4];
#pragma unroll
    for (int nn = 0; nn < 4; ++nn) {       // lane holds col n, K = kh*8..+7, 16+kh*8..+7
      const int bn = (wc << 6) + (nn << 4) + lm;
      const __bf16* bp = Bs + bn * 40 + kh * 8;
      bfg[nn] = frag_cat(bp, bp + 16);
    }
#pragma unroll
    for (int mm = 0; mm < 2; ++mm)
#pragma unroll
      for (int nn = 0; nn < 4; ++nn)
        acc[mm][nn] = __builtin_amdgcn_wmma_f32_16x16x32_bf16(
            false, af[mm], false, bfg[nn], (short)0, acc[mm][nn], false, false);

    if (kt + 1 < nk) stage(cur ^ 1);
    __syncthreads();
    cur ^= 1;
  }

  // epilogue: C layout VGPR r -> row r (+8 for upper lane half), col = lane&15
#pragma unroll
  for (int mm = 0; mm < 2; ++mm) {
#pragma unroll
    for (int nn = 0; nn < 4; ++nn) {
      const int col = nbase + (wc << 6) + (nn << 4) + lm;
#pragma unroll
      for (int r = 0; r < 8; ++r) {
        const int row = mbase + (wr << 5) + (mm << 4) + (kh << 3) + r;
        if (row < g.M && col < g.N) {
          float c = acc[mm][nn][r];
          if (g.bias) c += g.bias[col];
          if (g.act == 1) c = fmaxf(c, 0.f);
          else if (g.act == 2) c = gelu_exact(c);
          if (g.add) c += g.add[(size_t)row * g.ldc + col];
          g.C[(size_t)row * g.ldc + col] = c;
        }
      }
    }
  }
}

// ---------------------------------------------------------------- small kernels
__global__ void k_copy(const float* s, float* d, int n) {
  int i = blockIdx.x * 256 + threadIdx.x; if (i < n) d[i] = s[i];
}

__global__ void k_build_ee(const float* adj, const float* nodes, float* x0) {
  int idx = blockIdx.x * 256 + threadIdx.x;            // BS*128
  if (idx >= 4096 * 128) return;
  int bs = idx >> 7, c = idx & 127;
  int b = bs >> 10, ij = bs & 1023, i = ij >> 5, j = ij & 31;
  float a = adj[(b << 10) + ij];
  float v = (c < 64) ? nodes[((b << 5) + i) * 64 + c] : nodes[((b << 5) + j) * 64 + (c - 64)];
  x0[idx] = v * a;
}

__global__ void k_broadcast(const float* h, float* mb) {
  int idx = blockIdx.x * 256 + threadIdx.x;            // BS*512
  int b = idx >> 19, e = idx & 511;
  mb[idx] = h[(b << 9) + e];
}

__global__ void k_layernorm(const float* x, const float* gg, const float* bb, float* out, int L) {
  int row = blockIdx.x, t = threadIdx.x;
  const float* xr = x + (size_t)row * L;
  __shared__ float s1[128], s2[128];
  float a = 0.f, b = 0.f;
  for (int e = t; e < L; e += 128) { float v = xr[e]; a += v; b += v * v; }
  s1[t] = a; s2[t] = b; __syncthreads();
  for (int o = 64; o > 0; o >>= 1) { if (t < o) { s1[t] += s1[t + o]; s2[t] += s2[t + o]; } __syncthreads(); }
  float mean = s1[0] / (float)L;
  float var = s2[0] / (float)L - mean * mean;
  float inv = rsqrtf(var + 1e-5f);
  for (int e = t; e < L; e += 128) out[(size_t)row * L + e] = (xr[e] - mean) * inv * gg[e] + bb[e];
}

__global__ void k_normqk(float* qkv) {                 // rows BS, ld 1536
  int row = blockIdx.x, t = threadIdx.x;
  float* p = qkv + (size_t)row * 1536;
  __shared__ float s1[128], s2[128];
  float q = 0.f, k = 0.f;
  for (int e = t; e < 512; e += 128) { float a = p[e]; q += a * a; float c = p[512 + e]; k += c * c; }
  s1[t] = q; s2[t] = k; __syncthreads();
  for (int o = 64; o > 0; o >>= 1) { if (t < o) { s1[t] += s1[t + o]; s2[t] += s2[t + o]; } __syncthreads(); }
  float rq = rsqrtf(s1[0]), rk = rsqrtf(s2[0]);
  for (int e = t; e < 512; e += 128) { p[e] *= rq; p[512 + e] *= rk; }
}

__global__ void k_kvred(const float* qkv, float* kvs) { // kvs[b,e] = sum_s k*v
  int e = blockIdx.x * 256 + threadIdx.x;              // grid (2, 4)
  int b = blockIdx.y;
  const float* base = qkv + (size_t)b * 1024 * 1536;
  float acc = 0.f;
  for (int s = 0; s < 1024; ++s) {
    const float* r = base + (size_t)s * 1536;
    acc += r[512 + e] * r[1024 + e];
  }
  kvs[(b << 9) + e] = acc;
}

__global__ void k_smul(float* qkv, const float* kvs) { // q-slot *= kvs (hydra s)
  int idx = blockIdx.x * 256 + threadIdx.x;            // BS*512
  int bs = idx >> 9, e = idx & 511;
  qkv[(size_t)bs * 1536 + e] *= kvs[((bs >> 10) << 9) + e];
}

__global__ void k_heads_dot(const float* x, const float* wf, const float* bf_,
                            const float* wb, const float* bb_, const float* wst,
                            const float* bst, const float* wfl, const float* bfl, float* lg) {
  int row = blockIdx.x, t = threadIdx.x;
  const float* xr = x + (size_t)row * 512;
  float p0 = 0, p1 = 0, p2 = 0, p3 = 0;
  for (int e = t; e < 512; e += 128) {
    float v = xr[e];
    p0 += v * wf[e]; p1 += v * wb[e]; p2 += v * wst[e]; p3 += v * wfl[e];
  }
  __shared__ float sm[4][128];
  sm[0][t] = p0; sm[1][t] = p1; sm[2][t] = p2; sm[3][t] = p3; __syncthreads();
  for (int o = 64; o > 0; o >>= 1) {
    if (t < o) { for (int k = 0; k < 4; ++k) sm[k][t] += sm[k][t + o]; }
    __syncthreads();
  }
  if (t == 0) {
    lg[0 * 4096 + row] = sm[0][0] + bf_[0];
    lg[1 * 4096 + row] = sm[1][0] + bb_[0];
    lg[2 * 4096 + row] = sm[2][0] + bst[0];
    lg[3 * 4096 + row] = sm[3][0] + bfl[0];
  }
}

__global__ void k_softmax(const float* lg, const int* mask, float* out) {
  int b = blockIdx.x, t = threadIdx.x;                 // block 256
  __shared__ float sm[256];
  const float* row = lg + b * 1024;
  float mx = -3.0e38f;
  for (int s = t; s < 1024; s += 256) {
    float v = row[s]; if (mask && mask[b * 1024 + s] == 0) v = -1.0e9f;
    mx = fmaxf(mx, v);
  }
  sm[t] = mx; __syncthreads();
  for (int o = 128; o > 0; o >>= 1) { if (t < o) sm[t] = fmaxf(sm[t], sm[t + o]); __syncthreads(); }
  mx = sm[0]; __syncthreads();
  float sum = 0.f;
  for (int s = t; s < 1024; s += 256) {
    float v = row[s]; if (mask && mask[b * 1024 + s] == 0) v = -1.0e9f;
    sum += expf(v - mx);
  }
  sm[t] = sum; __syncthreads();
  for (int o = 128; o > 0; o >>= 1) { if (t < o) sm[t] += sm[t + o]; __syncthreads(); }
  float inv = 1.0f / sm[0];
  for (int s = t; s < 1024; s += 256) {
    float v = row[s]; if (mask && mask[b * 1024 + s] == 0) v = -1.0e9f;
    out[b * 1024 + s] = expf(v - mx) * inv;
  }
}

__global__ void k_stopflow(const float* lgs, const float* lgf, float* outs, float* outf) {
  int b = blockIdx.x, t = threadIdx.x;
  __shared__ float s1[256], s2[256];
  float a = 0.f, c = 0.f;
  for (int s = t; s < 1024; s += 256) { a += lgs[b * 1024 + s]; c += lgf[b * 1024 + s]; }
  s1[t] = a; s2[t] = c; __syncthreads();
  for (int o = 128; o > 0; o >>= 1) { if (t < o) { s1[t] += s1[t + o]; s2[t] += s2[t + o]; } __syncthreads(); }
  if (t == 0) { outs[b] = 1.0f / (1.0f + expf(-s1[0])); outf[b] = s2[0] * (1.0f / 1024.0f); }
}

__global__ void k_ne(const float* xnt, const float* adj, float* one) {
  int idx = blockIdx.x * 256 + threadIdx.x;            // B*N*D = 8192
  if (idx >= 4 * 32 * 64) return;
  int b = idx / 2048, r = idx % 2048, i = r / 64, d = r % 64;
  float s = 0.f, asum = 0.f;
  for (int j = 0; j < 32; ++j) {
    s += xnt[((size_t)(b * 1024 + i * 32 + j)) * 64 + d];
    asum += adj[b * 1024 + i * 32 + j];
  }
  one[idx] = (asum == 0.f) ? 0.f : s;
}

// ---------------------------------------------------------------- host side
struct TBp {
  const float *out_b, *out_w, *proj_b, *proj_w, *qkv_b, *qkv_w;
  const float *lin1_b, *lin1_w, *lin2_b, *lin2_w, *lin3_b, *lin3_w;
  const float *mix_b, *mix_w;
  const float *n1_b, *n1_g, *n2_b, *n2_g;
  const float *rp_b, *rp_w;
};

static TBp readTB(void* const* d_in, int& p, bool has_mix, bool has_rp) {
  TBp t{};
  auto F = [&]() { return (const float*)d_in[p++]; };
  t.out_b = F(); t.out_w = F(); t.proj_b = F(); t.proj_w = F(); t.qkv_b = F(); t.qkv_w = F();
  t.lin1_b = F(); t.lin1_w = F(); t.lin2_b = F(); t.lin2_w = F(); t.lin3_b = F(); t.lin3_w = F();
  if (has_mix) { t.mix_b = F(); t.mix_w = F(); }
  t.n1_b = F(); t.n1_g = F(); t.n2_b = F(); t.n2_g = F();
  if (has_rp) { t.rp_b = F(); t.rp_w = F(); }
  return t;
}

static inline void launch_gemm(hipStream_t st, const float* A, int lda, const float* W, int ldw,
                               const float* bias, const float* add, float* C, int ldc,
                               int M, int N, int K, int act) {
  GemmArgs g; g.A = A; g.W = W; g.bias = bias; g.add = add; g.C = C;
  g.M = M; g.N = N; g.K = K; g.lda = lda; g.ldw = ldw; g.ldc = ldc; g.act = act;
  dim3 grid((N + 127) / 128, (M + 127) / 128);
  hipLaunchKernelGGL(gemm_bf16_wmma, grid, dim3(256), 0, st, g);
}

// hydra attention: 8 heads, per-head QKV -> normalize q/k -> kv reduce ->
// s = q * kv -> out GEMM -> per-head accumulated proj GEMM.
static void run_attention(hipStream_t st, const float* x, const TBp& t, int outd,
                          float* qkvh, float* kvs, float* hoh, float* po) {
  for (int h = 0; h < 8; ++h) {
    launch_gemm(st, x, 512, t.qkv_w + (size_t)h * 512 * 1536, 1536,
                t.qkv_b + (size_t)h * 1536, nullptr, qkvh, 1536, 4096, 1536, 512, 0);
    hipLaunchKernelGGL(k_normqk, dim3(4096), dim3(128), 0, st, qkvh);
    hipLaunchKernelGGL(k_kvred, dim3(2, 4), dim3(256), 0, st, (const float*)qkvh, kvs);
    hipLaunchKernelGGL(k_smul, dim3(4096 * 512 / 256), dim3(256), 0, st, qkvh, (const float*)kvs);
    launch_gemm(st, qkvh, 1536, t.out_w + (size_t)h * 512 * outd, outd,
                t.out_b + (size_t)h * outd, nullptr, hoh, outd, 4096, outd, 512, 0);
    launch_gemm(st, hoh, outd, t.proj_w + (size_t)h * outd * outd, outd,
                h == 0 ? t.proj_b : nullptr, h == 0 ? nullptr : (const float*)po,
                po, outd, 4096, outd, outd, 0);
  }
}

extern "C" void kernel_launch(void* const* d_in, const int* in_sizes, int n_in,
                              void* d_out, int out_size, void* d_ws, size_t ws_size,
                              hipStream_t stream) {
  (void)in_sizes; (void)n_in; (void)out_size; (void)ws_size;
  const float* adj   = (const float*)d_in[0];
  const float* nodes = (const float*)d_in[1];
  const float* obs   = (const float*)d_in[2];
  const int*   mask  = (const int*)d_in[3];

  // params flattened in jax pytree (sorted-key) order
  int p = 4;
  auto F = [&]() { return (const float*)d_in[p++]; };
  const float* bck_b = F(); const float* bck_w = F();
  TBp blk[4];
  for (int i = 0; i < 4; ++i) blk[i] = readTB(d_in, p, /*mix=*/true, /*rp=*/i == 0);
  const float* flow_b = F(); const float* flow_w = F();
  const float* fwd_b  = F(); const float* fwd_w  = F();
  TBp nt = readTB(d_in, p, /*mix=*/false, /*rp=*/true);
  const float* ob_b0 = F(); const float* ob_w0 = F();
  const float* ob_b1 = F(); const float* ob_w1 = F();
  const float* ob_b2 = F(); const float* ob_w2 = F();
  const float* stop_b = F(); const float* stop_w = F();

  // workspace carve-up: ~18.9M floats (~76 MB) total
  float* ws   = (float*)d_ws;
  float* xa   = ws;                    // BS x 512
  float* xb   = ws + 2097152;          // BS x 512
  float* res  = ws + 4194304;          // BS x 512
  float* mb   = ws + 6291456;          // BS x 512 (mix broadcast / temp)
  float* qkvh = ws + 8388608;          // BS x 1536
  float* hoh  = ws + 14680064;         // BS x 512
  float* po   = ws + 16777216;         // BS x 512
  float* kvs  = ws + 18874368;         // 4 x 512
  float* ha   = ws + 18876416;         // 4 x 512
  float* hb   = ws + 18878464;         // 4 x 512
  float* lg   = ws + 18880512;         // 4 x 4096
  float* out  = (float*)d_out;         // [bck 4096 | fwd 4096 | stop 4 | flow 4 | ne 8192]

  // ---- obs MLP (M=4) + broadcast to mix input
  launch_gemm(stream, obs, 1024, ob_w0, 512, ob_b0, nullptr, ha, 512, 4, 512, 1024, 1);
  launch_gemm(stream, ha, 512, ob_w1, 512, ob_b1, nullptr, hb, 512, 4, 512, 512, 1);
  launch_gemm(stream, hb, 512, ob_w2, 512, ob_b2, nullptr, ha, 512, 4, 512, 512, 1);
  hipLaunchKernelGGL(k_broadcast, dim3(4096 * 512 / 256), dim3(256), 0, stream, (const float*)ha, mb);

  // ---- edge embeddings: (BS,128) gated by adj
  hipLaunchKernelGGL(k_build_ee, dim3(4096 * 128 / 256), dim3(256), 0, stream, adj, nodes, xa);

  // ---- transformer trunk
  for (int i = 0; i < 4; ++i) {
    const TBp& t = blk[i];
    const int din = (i == 0) ? 128 : 512;
    // residual path
    if (t.rp_w)
      launch_gemm(stream, xa, din, t.rp_w, 512, t.rp_b, nullptr, res, 512, 4096, 512, din, 0);
    else
      hipLaunchKernelGGL(k_copy, dim3(4096 * 512 / 256), dim3(256), 0, stream, (const float*)xa, res, 4096 * 512);
    // mix: concat(x, m) @ W  == x@W_top + m@W_bot
    launch_gemm(stream, xa, din, t.mix_w, din, t.mix_b, nullptr, xb, din, 4096, din, din, 0);
    launch_gemm(stream, mb, 512, t.mix_w + (size_t)din * din, din, nullptr, (const float*)xb, xb, din, 4096, din, 512, 0);
    // lin1 + gelu, ln1, lin2 + gelu
    launch_gemm(stream, xb, din, t.lin1_w, 512, t.lin1_b, nullptr, xa, 512, 4096, 512, din, 2);
    hipLaunchKernelGGL(k_layernorm, dim3(4096), dim3(128), 0, stream, (const float*)xa, t.n1_g, t.n1_b, xb, 512);
    launch_gemm(stream, xb, 512, t.lin2_w, 512, t.lin2_b, nullptr, xa, 512, 4096, 512, 512, 2);
    // hydra attention -> po
    run_attention(stream, xa, t, 512, qkvh, kvs, hoh, po);
    // ln2, lin3 + gelu + residual
    hipLaunchKernelGGL(k_layernorm, dim3(4096), dim3(128), 0, stream, (const float*)po, t.n2_g, t.n2_b, xb, 512);
    launch_gemm(stream, xb, 512, t.lin3_w, 512, t.lin3_b, (const float*)res, xa, 512, 4096, 512, 512, 2);
  }

  // ---- heads on final ee (= xa)
  hipLaunchKernelGGL(k_heads_dot, dim3(4096), dim3(128), 0, stream, (const float*)xa,
                     fwd_w, fwd_b, bck_w, bck_b, stop_w, stop_b, flow_w, flow_b, lg);
  hipLaunchKernelGGL(k_softmax, dim3(4), dim3(256), 0, stream, (const float*)(lg + 1 * 4096), (const int*)nullptr, out);        // bck
  hipLaunchKernelGGL(k_softmax, dim3(4), dim3(256), 0, stream, (const float*)(lg + 0 * 4096), mask, out + 4096);               // fwd
  hipLaunchKernelGGL(k_stopflow, dim3(4), dim3(256), 0, stream, (const float*)(lg + 2 * 4096), (const float*)(lg + 3 * 4096),
                     out + 8192, out + 8196);

  // ---- node_tb head (din=512, out=64, rp, no mix); input ee stays in xa
  launch_gemm(stream, xa, 512, nt.rp_w, 64, nt.rp_b, nullptr, res, 64, 4096, 64, 512, 0);
  launch_gemm(stream, xa, 512, nt.lin1_w, 512, nt.lin1_b, nullptr, xb, 512, 4096, 512, 512, 2);
  hipLaunchKernelGGL(k_layernorm, dim3(4096), dim3(128), 0, stream, (const float*)xb, nt.n1_g, nt.n1_b, mb, 512);
  launch_gemm(stream, mb, 512, nt.lin2_w, 512, nt.lin2_b, nullptr, xb, 512, 4096, 512, 512, 2);
  run_attention(stream, xb, nt, 64, qkvh, kvs, hoh, po);
  hipLaunchKernelGGL(k_layernorm, dim3(4096), dim3(128), 0, stream, (const float*)po, nt.n2_g, nt.n2_b, hoh, 64);
  launch_gemm(stream, hoh, 64, nt.lin3_w, 64, nt.lin3_b, (const float*)res, xb, 64, 4096, 64, 64, 2);
  hipLaunchKernelGGL(k_ne, dim3((8192 + 255) / 256), dim3(256), 0, stream, (const float*)xb, adj, out + 8200);
}
